// ScaledTernaryPackedMM_90434831385366
// MI455X (gfx1250) — matrix operand order repaired and checked by reading, not compiled, not run on hardware
//
#include <hip/hip_runtime.h>

typedef __attribute__((ext_vector_type(8)))  float   v8f;
typedef __attribute__((ext_vector_type(8)))  __bf16  v8bf;
typedef __attribute__((ext_vector_type(16))) __bf16  v16bf;
typedef __attribute__((ext_vector_type(2)))  __bf16  v2bf;

#define D_IN   4096
#define UNITS  4096
#define BM     128
#define BN     128
#define BK     32
#define LDS_STRIDE 40   // bf16 elems per row: 32 data + 8 pad = 80 bytes (16B aligned)

union V16 { v16bf v; v8bf h[2]; };

// ---------- helpers ----------
__device__ __forceinline__ unsigned cvt2(float a, float b) {
  v2bf t; t[0] = (__bf16)a; t[1] = (__bf16)b;
  union { v2bf v; unsigned u; } cv; cv.v = t;
  return cv.u;
}

// field -> bf16 bits: 0 -> -1 (0xBF80), 1 -> 0, 2 -> 1 (0x3F80), 3 -> 2 (0x4000)
__device__ __forceinline__ uint2 decByte(int v) {
  const unsigned long long LUT = 0x40003F800000BF80ULL;
  unsigned w0 = (unsigned short)(LUT >> (((v     ) & 3) * 16));
  unsigned w1 = (unsigned short)(LUT >> (((v >> 2) & 3) * 16));
  unsigned w2 = (unsigned short)(LUT >> (((v >> 4) & 3) * 16));
  unsigned w3 = (unsigned short)(LUT >> (((v >> 6) & 3) * 16));
  uint2 r; r.x = w0 | (w1 << 16); r.y = w2 | (w3 << 16);
  return r;
}

// ---------- async global->LDS path (CDNA5), with guards ----------
#if defined(__gfx1250__) && __has_builtin(__builtin_amdgcn_global_load_async_to_lds_b128)
#define USE_ASYNC_LDS 1
typedef __attribute__((ext_vector_type(4))) int i4vec;
typedef __attribute__((address_space(1))) i4vec* GP4;   // global int4*
typedef __attribute__((address_space(3))) i4vec* LP4;   // LDS int4*
#if __has_builtin(__builtin_amdgcn_s_wait_asynccnt)
#define WAIT_COPY() __builtin_amdgcn_s_wait_asynccnt(0)
#else
#define WAIT_COPY() asm volatile("s_wait_asynccnt 0x0" ::: "memory")
#endif
#else
#define USE_ASYNC_LDS 0
#define WAIT_COPY()
#endif

// ---------- pre-pass 1: X fp32 -> bf16 ----------
__global__ __launch_bounds__(256, 1)
void ConvertXToBf16(const float* __restrict__ x, unsigned short* __restrict__ xbf,
                    long long n8) {
  long long idx = (long long)blockIdx.x * 256 + threadIdx.x;
  if (idx >= n8) return;
  const float4* src = (const float4*)(x + idx * 8);
  float4 a = src[0], b = src[1];
  uint4 p;
  p.x = cvt2(a.x, a.y); p.y = cvt2(a.z, a.w);
  p.z = cvt2(b.x, b.y); p.w = cvt2(b.z, b.w);
  *(uint4*)(xbf + idx * 8) = p;
}

// ---------- pre-pass 2: packed ternary -> bf16 ----------
__global__ __launch_bounds__(256, 1)
void DecodeWToBf16(const int* __restrict__ pw, unsigned short* __restrict__ wbf,
                   long long n4) {
  long long idx = (long long)blockIdx.x * 256 + threadIdx.x;
  if (idx >= n4) return;
  int4 v = *(const int4*)(pw + idx * 4);
  uint2 d0 = decByte(v.x), d1 = decByte(v.y), d2 = decByte(v.z), d3 = decByte(v.w);
  unsigned short* dst = wbf + idx * 16;
  uint4 q0; q0.x = d0.x; q0.y = d0.y; q0.z = d1.x; q0.w = d1.y;
  uint4 q1; q1.x = d2.x; q1.y = d2.y; q1.z = d3.x; q1.w = d3.y;
  *(uint4*)(dst)     = q0;
  *(uint4*)(dst + 8) = q1;
}

// ---------- main GEMM: bf16 WMMA, double-buffered LDS ----------
__global__ __launch_bounds__(256, 1)
void GemmBf16Wmma(const unsigned short* __restrict__ xbf,
                  const unsigned short* __restrict__ wbf,
                  const float* __restrict__ scale,
                  const float* __restrict__ bias,
                  float* __restrict__ out) {
  __shared__ __align__(16) unsigned short lA[2][BM * LDS_STRIDE];
  __shared__ __align__(16) unsigned short lB[2][BN * LDS_STRIDE];

  const int tid  = threadIdx.x;
  const int lane = tid & 31;
  const int wave = tid >> 5;
  const int wm   = wave & 3;     // 4 row-groups of 32 rows
  const int wn   = wave >> 2;    // 2 col-groups of 64 cols
  const int hi   = lane >> 4;
  const int l16  = lane & 15;

  const int blockN = blockIdx.x;
  const int blockM = blockIdx.y;

  // cooperative-copy mapping: thread t covers tile row t/2, k-half t%2 (16 bf16)
  const int row  = tid >> 1;
  const int half = tid & 1;
  const unsigned short* gA = xbf + (size_t)(blockM * BM + row) * D_IN + half * 16;
  const unsigned short* gB = wbf + (size_t)(blockN * BN + row) * D_IN + half * 16;
  const int sOff = row * LDS_STRIDE + half * 16;

#if USE_ASYNC_LDS
#define ISSUE_COPY(bufi, kb) do {                                                                     \
    __builtin_amdgcn_global_load_async_to_lds_b128((GP4)(gA + (kb)),     (LP4)&lA[bufi][sOff],     0, 0); \
    __builtin_amdgcn_global_load_async_to_lds_b128((GP4)(gA + (kb) + 8), (LP4)&lA[bufi][sOff + 8], 0, 0); \
    __builtin_amdgcn_global_load_async_to_lds_b128((GP4)(gB + (kb)),     (LP4)&lB[bufi][sOff],     0, 0); \
    __builtin_amdgcn_global_load_async_to_lds_b128((GP4)(gB + (kb) + 8), (LP4)&lB[bufi][sOff + 8], 0, 0); \
  } while (0)
#else
#define ISSUE_COPY(bufi, kb) do {                                                            \
    uint4 a0 = *(const uint4*)(gA + (kb)); uint4 a1 = *(const uint4*)(gA + (kb) + 8);        \
    uint4 b0 = *(const uint4*)(gB + (kb)); uint4 b1 = *(const uint4*)(gB + (kb) + 8);        \
    *(uint4*)&lA[bufi][sOff] = a0; *(uint4*)&lA[bufi][sOff + 8] = a1;                        \
    *(uint4*)&lB[bufi][sOff] = b0; *(uint4*)&lB[bufi][sOff + 8] = b1;                        \
  } while (0)
#endif

  v8f zero = {};
  v8f acc[2][4];
#pragma unroll
  for (int m = 0; m < 2; ++m)
#pragma unroll
    for (int n = 0; n < 4; ++n)
      acc[m][n] = zero;

  ISSUE_COPY(0, 0);
  int buf = 0;

  for (int kb = 0; kb < D_IN; kb += BK) {
    WAIT_COPY();          // own async copies for `buf` complete (no-op on sync path)
    __syncthreads();      // all waves' copies visible; all prior reads of buf^1 done
    if (kb + BK < D_IN) ISSUE_COPY(buf ^ 1, kb + BK);

    // A fragments (16x32 bf16): lanes<16: v0..7=K0..15;
    //                           lanes>=16: v0..7=K16..31
    V16 afrag[2];
#pragma unroll
    for (int m = 0; m < 2; ++m) {
      const unsigned short* ap = &lA[buf][(wm * 32 + m * 16 + l16) * LDS_STRIDE];
      afrag[m].h[0] = *(const v8bf*)(ap + hi * 16);
      afrag[m].h[1] = *(const v8bf*)(ap + hi * 16 + 8);
    }
    // B fragments (32x16 bf16): lanes<16 hold K0..15, lanes>=16 hold K16..31
#pragma unroll
    for (int n = 0; n < 4; ++n) {
      const unsigned short* bp = &lB[buf][(wn * 64 + n * 16 + l16) * LDS_STRIDE + hi * 16];
      V16 bfrag;
      bfrag.h[0] = *(const v8bf*)(bp);
      bfrag.h[1] = *(const v8bf*)(bp + 8);
#pragma unroll
      for (int m = 0; m < 2; ++m) {
        acc[m][n] = __builtin_amdgcn_wmma_f32_16x16x32_bf16(
            false, afrag[m].v, false, bfrag.v, (short)0, acc[m][n], false, false);
      }
    }
    buf ^= 1;
  }

  // epilogue: scale, bias, clip, store
  const int colBase = blockN * BN + wn * 64;
  const int rowBase = blockM * BM + wm * 32;
#pragma unroll
  for (int n = 0; n < 4; ++n) {
    const int col = colBase + n * 16 + l16;
    const float sc = scale[col];
    const float bi = bias[col];
#pragma unroll
    for (int m = 0; m < 2; ++m) {
      const int r0 = rowBase + m * 16 + hi * 8;
#pragma unroll
      for (int j = 0; j < 8; ++j) {
        float v = acc[m][n][j] * sc + bi;
        v = fminf(fmaxf(v, -100.0f), 100.0f);
        out[(size_t)(r0 + j) * UNITS + col] = v;
      }
    }
  }
#undef ISSUE_COPY
}

// ---------- fused fallback (used only if workspace is too small) ----------
__global__ __launch_bounds__(256, 1)
void FusedTernaryWmma(const float* __restrict__ x,
                      const int*   __restrict__ pw,
                      const float* __restrict__ scale,
                      const float* __restrict__ bias,
                      float*       __restrict__ out) {
  __shared__ __align__(16) unsigned short lA[BM * LDS_STRIDE];
  __shared__ __align__(16) unsigned short lB[BN * LDS_STRIDE];

  const int tid  = threadIdx.x;
  const int lane = tid & 31;
  const int wave = tid >> 5;
  const int wm   = wave & 3;
  const int wn   = wave >> 2;
  const int hi   = lane >> 4;
  const int l16  = lane & 15;
  const int blockN = blockIdx.x;
  const int blockM = blockIdx.y;
  const int ldRow  = tid >> 1;
  const int ldHalf = tid & 1;

  const float* xRow = x  + (size_t)(blockM * BM + ldRow) * D_IN       + ldHalf * 16;
  const int*   pRow = pw + (size_t)(blockN * BN + ldRow) * (D_IN / 4) + ldHalf * 4;

  v8f zero = {};
  v8f acc[2][4];
#pragma unroll
  for (int m = 0; m < 2; ++m)
#pragma unroll
    for (int n = 0; n < 4; ++n)
      acc[m][n] = zero;

  for (int kb = 0; kb < D_IN; kb += BK) {
    const float4* xg = (const float4*)(xRow + kb);
    float4 f0 = xg[0], f1 = xg[1], f2 = xg[2], f3 = xg[3];
    uint4 pA0, pA1;
    pA0.x = cvt2(f0.x, f0.y); pA0.y = cvt2(f0.z, f0.w);
    pA0.z = cvt2(f1.x, f1.y); pA0.w = cvt2(f1.z, f1.w);
    pA1.x = cvt2(f2.x, f2.y); pA1.y = cvt2(f2.z, f2.w);
    pA1.z = cvt2(f3.x, f3.y); pA1.w = cvt2(f3.z, f3.w);

    const int4 pv = *(const int4*)(pRow + (kb >> 2));
    uint2 d0 = decByte(pv.x), d1 = decByte(pv.y), d2 = decByte(pv.z), d3 = decByte(pv.w);
    uint4 pB0, pB1;
    pB0.x = d0.x; pB0.y = d0.y; pB0.z = d1.x; pB0.w = d1.y;
    pB1.x = d2.x; pB1.y = d2.y; pB1.z = d3.x; pB1.w = d3.y;

    unsigned short* aDst = &lA[ldRow * LDS_STRIDE + ldHalf * 16];
    unsigned short* bDst = &lB[ldRow * LDS_STRIDE + ldHalf * 16];
    *(uint4*)(aDst)     = pA0;
    *(uint4*)(aDst + 8) = pA1;
    *(uint4*)(bDst)     = pB0;
    *(uint4*)(bDst + 8) = pB1;

    __syncthreads();

    V16 afrag[2];
#pragma unroll
    for (int m = 0; m < 2; ++m) {
      const unsigned short* ap = &lA[(wm * 32 + m * 16 + l16) * LDS_STRIDE];
      afrag[m].h[0] = *(const v8bf*)(ap + hi * 16);
      afrag[m].h[1] = *(const v8bf*)(ap + hi * 16 + 8);
    }
#pragma unroll
    for (int n = 0; n < 4; ++n) {
      const unsigned short* bp = &lB[(wn * 64 + n * 16 + l16) * LDS_STRIDE + hi * 16];
      V16 bfrag;
      bfrag.h[0] = *(const v8bf*)(bp);
      bfrag.h[1] = *(const v8bf*)(bp + 8);
#pragma unroll
      for (int m = 0; m < 2; ++m) {
        acc[m][n] = __builtin_amdgcn_wmma_f32_16x16x32_bf16(
            false, afrag[m].v, false, bfrag.v, (short)0, acc[m][n], false, false);
      }
    }
    __syncthreads();
  }

  const int colBase = blockN * BN + wn * 64;
  const int rowBase = blockM * BM + wm * 32;
#pragma unroll
  for (int n = 0; n < 4; ++n) {
    const int col = colBase + n * 16 + l16;
    const float sc = scale[col];
    const float bi = bias[col];
#pragma unroll
    for (int m = 0; m < 2; ++m) {
      const int r0 = rowBase + m * 16 + hi * 8;
#pragma unroll
      for (int j = 0; j < 8; ++j) {
        float v = acc[m][n][j] * sc + bi;
        v = fminf(fmaxf(v, -100.0f), 100.0f);
        out[(size_t)(r0 + j) * UNITS + col] = v;
      }
    }
  }
}

extern "C" void kernel_launch(void* const* d_in, const int* in_sizes, int n_in,
                              void* d_out, int out_size, void* d_ws, size_t ws_size,
                              hipStream_t stream) {
  const float* x     = (const float*)d_in[0];
  const int*   pw    = (const int*)d_in[1];
  const float* scale = (const float*)d_in[2];
  const float* bias  = (const float*)d_in[3];
  float*       out   = (float*)d_out;

  const int rows = in_sizes[0] / D_IN;   // 8192
  const size_t needX = (size_t)rows  * D_IN * sizeof(unsigned short); // 64 MB
  const size_t needW = (size_t)UNITS * D_IN * sizeof(unsigned short); // 32 MB

  if (ws_size >= needX + needW) {
    unsigned short* xbf = (unsigned short*)d_ws;
    unsigned short* wbf = (unsigned short*)((char*)d_ws + needX);
    const long long n8 = (long long)rows * D_IN / 8;
    const long long n4 = (long long)UNITS * (D_IN / 4) / 4;
    hipLaunchKernelGGL(ConvertXToBf16, dim3((unsigned)((n8 + 255) / 256)), dim3(256), 0, stream,
                       x, xbf, n8);
    hipLaunchKernelGGL(DecodeWToBf16, dim3((unsigned)((n4 + 255) / 256)), dim3(256), 0, stream,
                       pw, wbf, n4);
    hipLaunchKernelGGL(GemmBf16Wmma, dim3(UNITS / BN, rows / BM), dim3(256), 0, stream,
                       xbf, wbf, scale, bias, out);
  } else {
    hipLaunchKernelGGL(FusedTernaryWmma, dim3(UNITS / BN, rows / BM), dim3(256), 0, stream,
                       x, pw, scale, bias, out);
  }
}
